// Int8SymmetricLinear_14602888806542
// MI455X (gfx1250) — matrix operand order repaired and checked by reading, not compiled, not run on hardware
//
#include <hip/hip_runtime.h>

// Int8-symmetric linear as f16-WMMA GEMM with f32 accumulation.
//   out[M,N] = X[M,K] x (W[N,K] * scale[N])^T + bias[N]
// M=8192, N=11008, K=4096; all tile sizes divide exactly.
//
// Fast path (needs ws >= (N*K + M*K)*2 bytes):
//   1) cvt kernels: W int8 -> f16 (exact), X f32 -> f16, into d_ws
//   2) gemm_f16: LDS double-buffered WMMA GEMM; tiles staged by the
//      Tensor Data Mover (tensor_load_to_lds) when available, else
//      global_load_async_to_lds_b128, else ds_store staging.
// Fallback: fused kernel with in-loop conversion (round-1 design).

typedef _Float16     v8h  __attribute__((ext_vector_type(8)));
typedef _Float16     v16h __attribute__((ext_vector_type(16)));
typedef float        v8f  __attribute__((ext_vector_type(8)));
typedef float        v4f  __attribute__((ext_vector_type(4)));
typedef int          v4i  __attribute__((ext_vector_type(4)));
typedef int          v8i  __attribute__((ext_vector_type(8)));
typedef unsigned int v4u  __attribute__((ext_vector_type(4)));

#define BLOCK_M 128
#define BLOCK_N 128

union V16U { struct { v8h lo, hi; } p; v16h v; };

// ---------------------------------------------------------------------------
// Staging-path selection: TDM > async-b128 > ds_store
// ---------------------------------------------------------------------------
#if __has_builtin(__builtin_amdgcn_tensor_load_to_lds) && __has_builtin(__builtin_amdgcn_s_wait_tensorcnt)
#define STAGE_MODE 2
#pragma message("gfx1250: staging via tensor_load_to_lds (TDM, 6-arg form)")
#elif __has_builtin(__builtin_amdgcn_global_load_async_to_lds_b128)
#define STAGE_MODE 1
#pragma message("gfx1250: staging via global_load_async_to_lds_b128")
#else
#define STAGE_MODE 0
#pragma message("gfx1250: staging via ds_store (no async builtins)")
#endif

typedef __attribute__((address_space(1))) v4i* as1_v4i_p;
typedef __attribute__((address_space(3))) v4i* as3_v4i_p;

#if STAGE_MODE == 1
__device__ __forceinline__ void async_copy16(const void* g, void* l) {
    __builtin_amdgcn_global_load_async_to_lds_b128(
        (as1_v4i_p)g, (as3_v4i_p)l, 0, 0);
}
__device__ __forceinline__ void wait_async0() {
#if __has_builtin(__builtin_amdgcn_s_wait_asynccnt)
    __builtin_amdgcn_s_wait_asynccnt(0);
#else
    asm volatile("s_wait_asynccnt 0" ::: "memory");
#endif
}
#endif

#if STAGE_MODE == 2
// DS-relative LDS byte offset of a __shared__ object (flat->AS3 cast keeps
// the low 32 bits, which is exactly the ISA's flat->LDS truncation rule).
__device__ __forceinline__ unsigned lds_off(void* p) {
    return (unsigned)(unsigned long long)(as3_v4i_p)p;
}

// D# group1 for a 2-D [128 rows x 64 halfs] tile, row stride K halfs,
// LDS padding: 16 B after every 128 B (=> 144 B LDS row stride).
__device__ __forceinline__ v8i make_g1_tile(int K) {
    v8i g = (v8i){};
    g[0] = (int)((1u << 16)    // data_size = 2 bytes
               | (1u << 20)    // pad_enable
               | (4u << 22)    // pad_interval: 32 DWORDs (128 B)
               | (3u << 25));  // pad_amount:   4 DWORDs (16 B)
    g[1] = (int)(((unsigned)K & 0xffffu) << 16);        // tensor_dim0[15:0]
    g[2] = (int)((((unsigned)K >> 16) & 0xffffu)        // tensor_dim0[31:16]
               | (128u << 16));                         // tensor_dim1[15:0] = 128
    g[3] = (int)(64u << 16);                            // tile_dim0 = 64 elements
    g[4] = 128;                                         // tile_dim1 = 128 rows
    g[5] = K;                                           // tensor_dim0_stride[31:0]
    g[6] = 0;                                           // stride[47:32], dim1_stride lo
    g[7] = 0;
    return g;
}

__device__ __forceinline__ void tdm_load_2d(unsigned lds_addr, const void* gaddr, v8i g1) {
    const unsigned long long ga = (unsigned long long)(size_t)gaddr;
    v4u g0;
    g0[0] = 1u;                                          // count=1 (valid), user mode
    g0[1] = lds_addr;                                    // lds_addr
    g0[2] = (unsigned)ga;                                // global_addr[31:0]
    g0[3] = ((unsigned)(ga >> 32) & 0x01FFFFFFu)         // global_addr[56:32]
          | 0x80000000u;                                 // type=2 ("image")
    const v4i z4 = (v4i){};
    const v8i z8 = (v8i){};
    // 6-arg toolchain variant: (v4u, v8i, v4i, v4i, v8i, i32 cpol)
    __builtin_amdgcn_tensor_load_to_lds(g0, g1, z4, z4, z8, 0);
}
#endif

// ---------------------------------------------------------------------------
// One-shot conversion kernels (bandwidth-bound prologue)
// ---------------------------------------------------------------------------
__global__ __launch_bounds__(256)
void cvt_w_i8_f16(const signed char* __restrict__ W, _Float16* __restrict__ Wh) {
    const long long i = ((long long)blockIdx.x * 256 + threadIdx.x) * 16;
    union { v4i v; signed char c[16]; } u;
    u.v = *(const v4i*)(W + i);
    v8h h0, h1;
#pragma unroll
    for (int r = 0; r < 8; ++r) {
        h0[r] = (_Float16)u.c[r];
        h1[r] = (_Float16)u.c[8 + r];
    }
    *(v8h*)(Wh + i)     = h0;
    *(v8h*)(Wh + i + 8) = h1;
}

__global__ __launch_bounds__(256)
void cvt_x_f32_f16(const float* __restrict__ X, _Float16* __restrict__ Xh) {
    const long long i = ((long long)blockIdx.x * 256 + threadIdx.x) * 16;
    const v4f* xp = (const v4f*)(X + i);
    v4f a0 = xp[0], a1 = xp[1], a2 = xp[2], a3 = xp[3];
    v8h h0, h1;
#pragma unroll
    for (int r = 0; r < 4; ++r) {
        h0[r]     = (_Float16)a0[r];
        h0[r + 4] = (_Float16)a1[r];
        h1[r]     = (_Float16)a2[r];
        h1[r + 4] = (_Float16)a3[r];
    }
    *(v8h*)(Xh + i)     = h0;
    *(v8h*)(Xh + i + 8) = h1;
}

// ---------------------------------------------------------------------------
// Main GEMM: f16 x f16 -> f32, 128x128 block, 8 waves (2x4), wave tile 64x32,
// KTILE=64 (two WMMA K-chunks per LDS tile), double-buffered LDS.
// ---------------------------------------------------------------------------
#define KT2   64
#define LDSH2 72   // row stride in halfs: 64 data + 8 pad = 144 B (16B-aligned)

__global__ __launch_bounds__(256)
void gemm_f16_wmma(const _Float16* __restrict__ Xh,   // [M,K]
                   const _Float16* __restrict__ Wh,   // [N,K]
                   const float* __restrict__ Wscale,
                   const float* __restrict__ Bias,
                   float* __restrict__ Out,
                   int M, int N, int K)
{
    __shared__ _Float16 As[2][BLOCK_M * LDSH2];
    __shared__ _Float16 Bs[2][BLOCK_N * LDSH2];

    const int tid  = threadIdx.x;
    const int wave = tid >> 5;
    const int lane = tid & 31;
    const int wm   = wave >> 2;       // 0..1
    const int wn   = wave & 3;        // 0..3
    const int half = lane >> 4;
    const int lrow = lane & 15;

    const int m_base = blockIdx.y * BLOCK_M;
    const int n_base = blockIdx.x * BLOCK_N;

    const _Float16* gA = Xh + (size_t)m_base * K;
    const _Float16* gB = Wh + (size_t)n_base * K;

    v8f acc[4][2];
#pragma unroll
    for (int i = 0; i < 4; ++i)
#pragma unroll
        for (int j = 0; j < 2; ++j)
            acc[i][j] = (v8f){};

    const int nk = K / KT2;

#if STAGE_MODE == 2
    const v8i g1 = make_g1_tile(K);
    const unsigned ldsA0 = lds_off(&As[0][0]), ldsA1 = lds_off(&As[1][0]);
    const unsigned ldsB0 = lds_off(&Bs[0][0]), ldsB1 = lds_off(&Bs[1][0]);

    // ---- TDM prologue: stage tile 0 ----
    if (wave == 0) {
        tdm_load_2d(ldsA0, gA, g1);
        tdm_load_2d(ldsB0, gB, g1);
        __builtin_amdgcn_s_wait_tensorcnt(0);
    }
    __syncthreads();

    for (int kt = 0; kt < nk; ++kt) {
        const int cur = kt & 1;
        const int nxt = cur ^ 1;

        if (kt + 1 < nk && wave == 0) {
            const int kk = (kt + 1) * KT2;
            tdm_load_2d(nxt ? ldsA1 : ldsA0, gA + kk, g1);
            tdm_load_2d(nxt ? ldsB1 : ldsB0, gB + kk, g1);
        }
#elif STAGE_MODE == 1
    // ---- async prologue: stage tile 0 (8 chunks of 16 B per thread) ----
    {
#pragma unroll
        for (int s = 0; s < 4; ++s) {
            const int c = s * 256 + tid, row = c >> 3, seg = c & 7;
            async_copy16(gA + (size_t)row * K + seg * 8, &As[0][row * LDSH2 + seg * 8]);
            async_copy16(gB + (size_t)row * K + seg * 8, &Bs[0][row * LDSH2 + seg * 8]);
        }
        wait_async0();
    }
    __syncthreads();

    for (int kt = 0; kt < nk; ++kt) {
        const int cur = kt & 1;
        const int nxt = cur ^ 1;

        if (kt + 1 < nk) {
            const int kk = (kt + 1) * KT2;
#pragma unroll
            for (int s = 0; s < 4; ++s) {
                const int c = s * 256 + tid, row = c >> 3, seg = c & 7;
                async_copy16(gA + (size_t)row * K + kk + seg * 8,
                             &As[nxt][row * LDSH2 + seg * 8]);
                async_copy16(gB + (size_t)row * K + kk + seg * 8,
                             &Bs[nxt][row * LDSH2 + seg * 8]);
            }
        }
#else
    // ---- ds_store prologue ----
    {
#pragma unroll
        for (int s = 0; s < 4; ++s) {
            const int c = s * 256 + tid, row = c >> 3, seg = c & 7;
            *(v4i*)&As[0][row * LDSH2 + seg * 8] = *(const v4i*)(gA + (size_t)row * K + seg * 8);
            *(v4i*)&Bs[0][row * LDSH2 + seg * 8] = *(const v4i*)(gB + (size_t)row * K + seg * 8);
        }
    }
    __syncthreads();

    for (int kt = 0; kt < nk; ++kt) {
        const int cur = kt & 1;
        const int nxt = cur ^ 1;

        v4i ra[4], rb[4];
        if (kt + 1 < nk) {
            const int kk = (kt + 1) * KT2;
#pragma unroll
            for (int s = 0; s < 4; ++s) {
                const int c = s * 256 + tid, row = c >> 3, seg = c & 7;
                ra[s] = *(const v4i*)(gA + (size_t)row * K + kk + seg * 8);
                rb[s] = *(const v4i*)(gB + (size_t)row * K + kk + seg * 8);
            }
        }
#endif
        // ---- compute from LDS buffer `cur`: 2 K-chunks x 8 WMMAs ----
        const _Float16* Ab = As[cur];
        const _Float16* Bb = Bs[cur];
#pragma unroll
        for (int kc = 0; kc < 2; ++kc) {
            v16h afrag[4], bfrag[2];
#pragma unroll
            for (int i = 0; i < 4; ++i) {
                const _Float16* p = Ab + (wm * 64 + i * 16 + lrow) * LDSH2 + kc * 32 + half * 8;
                V16U u;
                u.p.lo = *(const v8h*)(p);       // K 0-7   / 8-15
                u.p.hi = *(const v8h*)(p + 16);  // K 16-23 / 24-31
                afrag[i] = u.v;
            }
#pragma unroll
            for (int j = 0; j < 2; ++j) {
                const _Float16* p = Bb + (wn * 32 + j * 16 + lrow) * LDSH2 + kc * 32 + half * 8;
                V16U u;
                u.p.lo = *(const v8h*)(p);       // K 0-7   / 8-15
                u.p.hi = *(const v8h*)(p + 16);  // K 16-23 / 24-31
                bfrag[j] = u.v;
            }
#pragma unroll
            for (int i = 0; i < 4; ++i)
#pragma unroll
                for (int j = 0; j < 2; ++j)
                    acc[i][j] = __builtin_amdgcn_wmma_f32_16x16x32_f16(
                        false, afrag[i], false, bfrag[j],
                        (short)0, acc[i][j], false, false);
        }

#if STAGE_MODE == 2
        if (kt + 1 < nk && wave == 0) __builtin_amdgcn_s_wait_tensorcnt(0);
#elif STAGE_MODE == 1
        if (kt + 1 < nk) wait_async0();
#else
        if (kt + 1 < nk) {
#pragma unroll
            for (int s = 0; s < 4; ++s) {
                const int c = s * 256 + tid, row = c >> 3, seg = c & 7;
                *(v4i*)&As[nxt][row * LDSH2 + seg * 8] = ra[s];
                *(v4i*)&Bs[nxt][row * LDSH2 + seg * 8] = rb[s];
            }
        }
#endif
        __syncthreads();
    }

    // ---- epilogue: per-channel scale + bias ----
#pragma unroll
    for (int j = 0; j < 2; ++j) {
        const int n_g = n_base + wn * 32 + j * 16 + lrow;
        const float sc = Wscale[n_g];
        const float bv = Bias[n_g];
#pragma unroll
        for (int i = 0; i < 4; ++i) {
            const int m0 = m_base + wm * 64 + i * 16 + half * 8;
#pragma unroll
            for (int r = 0; r < 8; ++r)
                Out[(size_t)(m0 + r) * N + n_g] = acc[i][j][r] * sc + bv;
        }
    }
}

// ---------------------------------------------------------------------------
// Fallback fused kernel (round-1 design): used only if ws is too small.
// ---------------------------------------------------------------------------
#define KTILE 32
#define LDSH  40

__device__ __forceinline__ void cvt_store_a(_Float16* dst, const v4f a[4]) {
#pragma unroll
    for (int q = 0; q < 2; ++q) {
        v8h h;
#pragma unroll
        for (int r = 0; r < 8; ++r)
            h[r] = (_Float16)a[q * 2 + (r >> 2)][r & 3];
        *(v8h*)(dst + q * 8) = h;
    }
}

__device__ __forceinline__ void cvt_store_b(_Float16* dst, v4i b) {
    union { v4i v; signed char c[16]; } u;
    u.v = b;
#pragma unroll
    for (int q = 0; q < 2; ++q) {
        v8h h;
#pragma unroll
        for (int r = 0; r < 8; ++r)
            h[r] = (_Float16)u.c[q * 8 + r];
        *(v8h*)(dst + q * 8) = h;
    }
}

__global__ __launch_bounds__(256)
void int8_linear_fused(const float* __restrict__ X,
                       const signed char* __restrict__ W,
                       const float* __restrict__ Wscale,
                       const float* __restrict__ Bias,
                       float* __restrict__ Out,
                       int M, int N, int K)
{
    __shared__ _Float16 As[2][BLOCK_M * LDSH];
    __shared__ _Float16 Bs[2][BLOCK_N * LDSH];

    const int tid  = threadIdx.x;
    const int wave = tid >> 5;
    const int lane = tid & 31;
    const int wm   = wave >> 2;
    const int wn   = wave & 3;
    const int half = lane >> 4;
    const int lrow = lane & 15;

    const int m_base = blockIdx.y * BLOCK_M;
    const int n_base = blockIdx.x * BLOCK_N;

    const int ldrow = tid >> 1;
    const int ldseg = (tid & 1) * 16;

    const float*       Xrow = X + (size_t)(m_base + ldrow) * K + ldseg;
    const signed char* Wrow = W + (size_t)(n_base + ldrow) * K + ldseg;

    v8f acc[4][2];
#pragma unroll
    for (int i = 0; i < 4; ++i)
#pragma unroll
        for (int j = 0; j < 2; ++j)
            acc[i][j] = (v8f){};

    {
        v4f a0[4];
        const v4f* xp = (const v4f*)Xrow;
#pragma unroll
        for (int q = 0; q < 4; ++q) a0[q] = xp[q];
        v4i b0 = *(const v4i*)Wrow;
        cvt_store_a(&As[0][ldrow * LDSH + ldseg], a0);
        cvt_store_b(&Bs[0][ldrow * LDSH + ldseg], b0);
    }
    __syncthreads();

    const int nk = K / KTILE;
    for (int kt = 0; kt < nk; ++kt) {
        const int cur = kt & 1;
        const int nxt = cur ^ 1;

        v4f an[4];
        v4i bn;
        if (kt + 1 < nk) {
            const int kk = (kt + 1) * KTILE;
            const v4f* xp = (const v4f*)(Xrow + kk);
#pragma unroll
            for (int q = 0; q < 4; ++q) an[q] = xp[q];
            bn = *(const v4i*)(Wrow + kk);
        }

        const _Float16* Ab = As[cur];
        const _Float16* Bb = Bs[cur];
        v16h afrag[4], bfrag[2];
#pragma unroll
        for (int i = 0; i < 4; ++i) {
            const _Float16* p = Ab + (wm * 64 + i * 16 + lrow) * LDSH + half * 8;
            V16U u;
            u.p.lo = *(const v8h*)(p);
            u.p.hi = *(const v8h*)(p + 16);
            afrag[i] = u.v;
        }
#pragma unroll
        for (int j = 0; j < 2; ++j) {
            const _Float16* p = Bb + (wn * 32 + j * 16 + lrow) * LDSH + half * 8;
            V16U u;
            u.p.lo = *(const v8h*)(p);
            u.p.hi = *(const v8h*)(p + 16);
            bfrag[j] = u.v;
        }

#pragma unroll
        for (int i = 0; i < 4; ++i)
#pragma unroll
            for (int j = 0; j < 2; ++j)
                acc[i][j] = __builtin_amdgcn_wmma_f32_16x16x32_f16(
                    false, afrag[i], false, bfrag[j],
                    (short)0, acc[i][j], false, false);

        if (kt + 1 < nk) {
            cvt_store_a(&As[nxt][ldrow * LDSH + ldseg], an);
            cvt_store_b(&Bs[nxt][ldrow * LDSH + ldseg], bn);
        }
        __syncthreads();
    }

#pragma unroll
    for (int j = 0; j < 2; ++j) {
        const int n_g = n_base + wn * 32 + j * 16 + lrow;
        const float sc = Wscale[n_g];
        const float bv = Bias[n_g];
#pragma unroll
        for (int i = 0; i < 4; ++i) {
            const int m0 = m_base + wm * 64 + i * 16 + half * 8;
#pragma unroll
            for (int r = 0; r < 8; ++r)
                Out[(size_t)(m0 + r) * N + n_g] = acc[i][j][r] * sc + bv;
        }
    }
}

// ---------------------------------------------------------------------------
// Host launcher
// ---------------------------------------------------------------------------
extern "C" void kernel_launch(void* const* d_in, const int* in_sizes, int n_in,
                              void* d_out, int out_size, void* d_ws, size_t ws_size,
                              hipStream_t stream) {
    (void)n_in; (void)out_size;

    const float*       x  = (const float*)d_in[0];
    const signed char* w  = (const signed char*)d_in[1];  // int8 weights [OUT, IN]
    const float*       sc = (const float*)d_in[2];        // [OUT, 1]
    const float*       b  = (const float*)d_in[3];        // [OUT]
    float*             out = (float*)d_out;

    const int OUT = in_sizes[2];              // 11008
    const int IN  = in_sizes[1] / OUT;        // 4096
    const int Mm  = in_sizes[0] / IN;         // 8192

    const size_t wElems = (size_t)OUT * IN;
    const size_t xElems = (size_t)Mm * IN;
    const size_t need   = (wElems + xElems) * sizeof(_Float16);

    dim3 grid(OUT / BLOCK_N, Mm / BLOCK_M);   // 86 x 64

    if (ws_size >= need) {
        _Float16* Wh = (_Float16*)d_ws;
        _Float16* Xh = Wh + wElems;
        cvt_w_i8_f16<<<(unsigned)(wElems / (256 * 16)), 256, 0, stream>>>(w, Wh);
        cvt_x_f32_f16<<<(unsigned)(xElems / (256 * 16)), 256, 0, stream>>>(x, Xh);
        gemm_f16_wmma<<<grid, 256, 0, stream>>>(Xh, Wh, sc, b, out, Mm, OUT, IN);
    } else {
        int8_linear_fused<<<grid, 256, 0, stream>>>(x, w, sc, b, out, Mm, OUT, IN);
    }
}
